// SketchEmbed_11381663334407
// MI455X (gfx1250) — matrix operand order repaired and checked by reading, not compiled, not run on hardware
//
#include <hip/hip_runtime.h>
#include <stdint.h>
#include <stddef.h>

// ---------------------------------------------------------------------------
// GATv2 triple-encoder + MLP head for MI455X (gfx1250, wave32, WMMA).
// GEMMs run on v_wmma_f32_16x16x32_bf16 (bf16 in, f32 accumulate).
// Edge softmax/aggregation is CSR-by-destination, one wave per node,
// register-resident softmax (no float atomics in hot path).
// ---------------------------------------------------------------------------

typedef __attribute__((ext_vector_type(16))) __bf16 v16bf;
typedef __attribute__((ext_vector_type(8)))  float  v8f;

union AFrag { v16bf v; float4 f4[2]; };

#define SLOPE 0.2f

// ---------------- CSR build ----------------
__global__ void k_deg_init(int* __restrict__ cnt, int n) {
  int i = blockIdx.x * blockDim.x + threadIdx.x;
  if (i < n) cnt[i] = 1;                       // implicit self loop
}
__global__ void k_deg_count(const int* __restrict__ dst, int e, int* __restrict__ cnt) {
  int i = blockIdx.x * blockDim.x + threadIdx.x;
  if (i < e) atomicAdd(&cnt[dst[i]], 1);
}
__global__ void __launch_bounds__(1024) k_scan(const int* __restrict__ cnt,
                                               int* __restrict__ rowptr, int n) {
  __shared__ int sh[1024];
  int t = threadIdx.x;
  int carry = 0;
  for (int base = 0; base < n; base += 1024) {
    int idx = base + t;
    int v = (idx < n) ? cnt[idx] : 0;
    sh[t] = v;
    __syncthreads();
    for (int off = 1; off < 1024; off <<= 1) {
      int x = (t >= off) ? sh[t - off] : 0;
      __syncthreads();
      sh[t] += x;
      __syncthreads();
    }
    if (idx < n) rowptr[idx] = carry + sh[t] - v;   // exclusive
    carry += sh[1023];
    __syncthreads();
  }
  if (t == 0) rowptr[n] = carry;
}
__global__ void k_fill_self(const int* __restrict__ rowptr, int* __restrict__ fillpos,
                            int* __restrict__ colidx, int n) {
  int i = blockIdx.x * blockDim.x + threadIdx.x;
  if (i < n) { int p = rowptr[i]; colidx[p] = i; fillpos[i] = p + 1; }
}
__global__ void k_fill_edges(const int* __restrict__ src, const int* __restrict__ dst, int e,
                             int* __restrict__ fillpos, int* __restrict__ colidx) {
  int i = blockIdx.x * blockDim.x + threadIdx.x;
  if (i < e) { int p = atomicAdd(&fillpos[dst[i]], 1); colidx[p] = src[i]; }
}

// ---------------- weight prep: f32 [K,256] -> bf16 transposed [256,K] ----------------
__global__ void k_prep_wt(const float* __restrict__ Wl, const float* __restrict__ Wr,
                          __bf16* __restrict__ Wlt, __bf16* __restrict__ Wrt, int K) {
  int i = blockIdx.x * blockDim.x + threadIdx.x;
  if (i >= 256 * K) return;
  int n = i / K, k = i - n * K;
  Wlt[i] = (__bf16)Wl[(size_t)k * 256 + n];
  Wrt[i] = (__bf16)Wr[(size_t)k * 256 + n];
}

// ---------------- WMMA GEMM: XL = X@Wl, XR = X@Wr (dout = 256) ----------------
// Block = 256 threads = 8 waves; block owns a 16-row tile of X (staged in LDS as bf16).
// Waves 0..3 -> XL cols [64w,64w+64); waves 4..7 -> XR.
__global__ void __launch_bounds__(256) k_gat_gemm(const float* __restrict__ X,
                                                  const __bf16* __restrict__ Wlt,
                                                  const __bf16* __restrict__ Wrt,
                                                  float* __restrict__ XL,
                                                  float* __restrict__ XR,
                                                  int K, int nRows) {
  __shared__ __attribute__((aligned(16))) __bf16 lds[16 * 264];   // K<=256, Kpad=K+8
  const int Kpad = K + 8;
  const int mBase = blockIdx.x * 16;

  for (int idx = threadIdx.x; idx < 16 * K; idx += 256) {
    int r = idx / K, c = idx - r * K;
    int row = mBase + r;
    lds[r * Kpad + c] = (row < nRows) ? (__bf16)X[(size_t)row * K + c] : (__bf16)0.0f;
  }
  __syncthreads();

  const int wid  = threadIdx.x >> 5;
  const int lane = threadIdx.x & 31;
  const __bf16* Wt = (wid < 4) ? Wlt : Wrt;
  float*        Ot = (wid < 4) ? XL  : XR;
  const int colBase = (wid & 3) * 64;
  const int half = lane >> 4;
  const int mr   = lane & 15;

  v8f c0 = {}, c1 = {}, c2 = {}, c3 = {};
  for (int k0 = 0; k0 < K; k0 += 32) {
    AFrag a;
    const __bf16* Arow = lds + mr * Kpad + k0 + 16 * half;
    a.f4[0] = *(const float4*)(Arow);
    a.f4[1] = *(const float4*)(Arow + 8);
    const __bf16* Wb = Wt + k0 + 16 * half;
    v16bf b0 = *(const v16bf*)(Wb + (size_t)(colBase +  0 + mr) * K);
    v16bf b1 = *(const v16bf*)(Wb + (size_t)(colBase + 16 + mr) * K);
    v16bf b2 = *(const v16bf*)(Wb + (size_t)(colBase + 32 + mr) * K);
    v16bf b3 = *(const v16bf*)(Wb + (size_t)(colBase + 48 + mr) * K);
    c0 = __builtin_amdgcn_wmma_f32_16x16x32_bf16(false, a.v, false, b0, (short)0, c0, false, false);
    c1 = __builtin_amdgcn_wmma_f32_16x16x32_bf16(false, a.v, false, b1, (short)0, c1, false, false);
    c2 = __builtin_amdgcn_wmma_f32_16x16x32_bf16(false, a.v, false, b2, (short)0, c2, false, false);
    c3 = __builtin_amdgcn_wmma_f32_16x16x32_bf16(false, a.v, false, b3, (short)0, c3, false, false);
  }

#pragma unroll
  for (int r = 0; r < 8; ++r) {
    int row = mBase + r + 8 * half;
    if (row < nRows) {
      size_t o = (size_t)row * 256 + colBase + mr;
      Ot[o +  0] = c0[r];
      Ot[o + 16] = c1[r];
      Ot[o + 32] = c2[r];
      Ot[o + 48] = c3[r];
    }
  }
}

// ---------------- per-node softmax + aggregation (one wave per dst node) ----------------
__global__ void __launch_bounds__(256) k_gat_aggregate(const float* __restrict__ XL,
                                                       const float* __restrict__ XR,
                                                       const float* __restrict__ att,
                                                       const float* __restrict__ bias,
                                                       const int* __restrict__ rowptr,
                                                       const int* __restrict__ colidx,
                                                       float* __restrict__ ebuf,
                                                       float* __restrict__ out,
                                                       int nNodes, int doRelu) {
  int wave = (int)((blockIdx.x * blockDim.x + threadIdx.x) >> 5);
  int lane = threadIdx.x & 31;
  if (wave >= nNodes) return;
  const int n  = wave;
  const int rp = rowptr[n];
  const int re = rowptr[n + 1];

  float xr[8], av[8];
#pragma unroll
  for (int j = 0; j < 8; ++j) {
    int d = lane + 32 * j;
    xr[j] = XR[(size_t)n * 256 + d];
    av[j] = att[d];
  }

  // pass 1: scores e_i = att . leaky_relu(xl[src] + xr[dst]); track max
  float m = -3.402823466e38f;
  for (int i = rp; i < re; ++i) {
    int s = colidx[i];
    if (i + 1 < re) __builtin_prefetch(XL + (size_t)colidx[i + 1] * 256, 0, 1);
    const float* xlr = XL + (size_t)s * 256;
    float t = 0.0f;
#pragma unroll
    for (int j = 0; j < 8; ++j) {
      float v = xlr[lane + 32 * j] + xr[j];
      v = (v > 0.0f) ? v : SLOPE * v;
      t += v * av[j];
    }
#pragma unroll
    for (int off = 16; off >= 1; off >>= 1) t += __shfl_xor(t, off, 32);
    ebuf[i] = t;                 // all lanes hold identical t; same-lane RAW keeps order
    m = fmaxf(m, t);
  }

  // pass 2: denominator
  float den = 0.0f;
  for (int i = rp; i < re; ++i) den += __expf(ebuf[i] - m);
  den += 1e-16f;

  // pass 3: weighted aggregation in registers
  float acc[8];
#pragma unroll
  for (int j = 0; j < 8; ++j) acc[j] = 0.0f;
  for (int i = rp; i < re; ++i) {
    int s = colidx[i];
    if (i + 1 < re) __builtin_prefetch(XL + (size_t)colidx[i + 1] * 256, 0, 1);
    float alpha = __expf(ebuf[i] - m) / den;
    const float* xlr = XL + (size_t)s * 256;
#pragma unroll
    for (int j = 0; j < 8; ++j) acc[j] += alpha * xlr[lane + 32 * j];
  }

#pragma unroll
  for (int j = 0; j < 8; ++j) {
    int d = lane + 32 * j;
    float v = acc[j] + bias[d];
    if (doRelu) v = fmaxf(v, 0.0f);
    out[(size_t)n * 256 + d] = v;
  }
}

// ---------------- deterministic mean pool (two stages, no atomics) ----------------
__global__ void k_pool_partial(const float* __restrict__ H, float* __restrict__ part, int n) {
  int d = threadIdx.x;                              // 256 threads
  float s = 0.0f;
  for (int r = blockIdx.x; r < n; r += gridDim.x) s += H[(size_t)r * 256 + d];
  part[(size_t)blockIdx.x * 256 + d] = s;
}
__global__ void k_pool_final(const float* __restrict__ part, float* __restrict__ emb,
                             int nPart, float invN) {
  int d = threadIdx.x;
  float s = 0.0f;
  for (int b = 0; b < nPart; ++b) s += part[(size_t)b * 256 + d];
  emb[d] = s * invN;
}

// ---------------- MLP head (tiny: 768 -> 256 -> 256 -> 256) ----------------
__global__ void __launch_bounds__(256) k_mlp(const float* __restrict__ emb,
                                             const float* __restrict__ W1, const float* __restrict__ b1,
                                             const float* __restrict__ W2, const float* __restrict__ b2,
                                             const float* __restrict__ W3, const float* __restrict__ b3,
                                             float* __restrict__ out) {
  __shared__ float se[768], h1[256], h2[256];
  int t = threadIdx.x;
  for (int i = t; i < 768; i += 256) se[i] = emb[i];
  __syncthreads();
  float s = b1[t];
  for (int i = 0; i < 768; ++i) s += se[i] * W1[i * 256 + t];
  h1[t] = fmaxf(s, 0.0f);
  __syncthreads();
  s = b2[t];
  for (int i = 0; i < 256; ++i) s += h1[i] * W2[i * 256 + t];
  h2[t] = fmaxf(s, 0.0f);
  __syncthreads();
  s = b3[t];
  for (int i = 0; i < 256; ++i) s += h2[i] * W3[i * 256 + t];
  out[t] = s;
}

// ===========================================================================
// Host orchestration
// ===========================================================================
struct EncP { const float *Wl[5], *Wr[5], *att[5], *b[5]; };   // exec order: in,h0,h1,h2,out

extern "C" void kernel_launch(void* const* d_in, const int* in_sizes, int n_in,
                              void* d_out, int out_size, void* d_ws, size_t ws_size,
                              hipStream_t stream) {
  const int Dd = 128;

  // encoder data inputs (dict order: lhs, rhs, sketch); emb slots: sketch=0, lhs=1, rhs=2
  const int xIdx[3] = {4, 0, 2};     // sketch, lhs, rhs
  const int eIdx[3] = {5, 1, 3};

  // ---- parse params (handle per-leaf flatten (n_in>=72) and concat (n_in==10)) ----
  EncP enc[3];
  const float *W1, *B1, *W2, *B2, *W3, *B3;
  if (n_in > 10) {
    const int base[3] = {46, 6, 26};         // sketch, lhs, rhs param leaf bases
    for (int s = 0; s < 3; ++s) {
      int b0 = base[s];
      int g_in = 0;
      for (int g = 0; g < 5; ++g)
        if (in_sizes[b0 + 4 * g] == Dd * 256) g_in = g;
      int order[5];
      if (g_in == 0) { order[0]=0; order[1]=1; order[2]=2; order[3]=3; order[4]=4; }          // insertion
      else           { order[0]=3; order[1]=0; order[2]=1; order[3]=2; order[4]=4; }          // sorted keys
      for (int l = 0; l < 5; ++l) {
        int g = order[l];
        enc[s].Wl[l]  = (const float*)d_in[b0 + 4 * g + 0];
        enc[s].Wr[l]  = (const float*)d_in[b0 + 4 * g + 1];
        enc[s].att[l] = (const float*)d_in[b0 + 4 * g + 2];
        enc[s].b[l]   = (const float*)d_in[b0 + 4 * g + 3];
      }
    }
    W1 = (const float*)d_in[66]; B1 = (const float*)d_in[67];
    W2 = (const float*)d_in[68]; B2 = (const float*)d_in[69];
    W3 = (const float*)d_in[70]; B3 = (const float*)d_in[71];
  } else {
    // concatenated pytree leaves, jax sorted-key dict order: h0,h1,h2,in,out
    const int pIdx[3] = {8, 6, 7};           // sketch, lhs, rhs
    for (int s = 0; s < 3; ++s) {
      const float* base = (const float*)d_in[pIdx[s]];
      const float* gp[5][4];
      const int Ks[5] = {256, 256, 256, 128, 256};
      size_t off = 0;
      for (int g = 0; g < 5; ++g) {
        gp[g][0] = base + off; off += (size_t)Ks[g] * 256;
        gp[g][1] = base + off; off += (size_t)Ks[g] * 256;
        gp[g][2] = base + off; off += 256;
        gp[g][3] = base + off; off += 256;
      }
      const int order[5] = {3, 0, 1, 2, 4};  // in,h0,h1,h2,out
      for (int l = 0; l < 5; ++l) {
        int g = order[l];
        enc[s].Wl[l] = gp[g][0]; enc[s].Wr[l] = gp[g][1];
        enc[s].att[l] = gp[g][2]; enc[s].b[l] = gp[g][3];
      }
    }
    const float* bb = (const float*)d_in[9];
    W1 = bb;            B1 = bb + 768 * 256;
    W2 = B1 + 256;      B2 = W2 + 256 * 256;
    W3 = B2 + 256;      B3 = W3 + 256 * 256;
  }

  // ---- workspace carve ----
  size_t maxN = 0, maxE = 0;
  for (int s = 0; s < 3; ++s) {
    size_t nn = (size_t)in_sizes[xIdx[s]] / Dd;
    size_t ee = (size_t)in_sizes[eIdx[s]] / 2;
    if (nn > maxN) maxN = nn;
    if (ee > maxE) maxE = ee;
  }
  size_t maxE2 = maxE + maxN;
  char* w = (char*)d_ws;
  auto alloc = [&](size_t bytes) -> void* {
    void* p = (void*)w;
    w += (bytes + 255) & ~(size_t)255;
    return p;
  };
  float* bufA    = (float*)alloc(maxN * 256 * sizeof(float));
  float* bufB    = (float*)alloc(maxN * 256 * sizeof(float));
  float* XL      = (float*)alloc(maxN * 256 * sizeof(float));
  float* XR      = (float*)alloc(maxN * 256 * sizeof(float));
  __bf16* Wlt    = (__bf16*)alloc(256 * 256 * sizeof(__bf16));
  __bf16* Wrt    = (__bf16*)alloc(256 * 256 * sizeof(__bf16));
  int*   rowptr  = (int*)alloc((maxN + 1) * sizeof(int));
  int*   fillpos = (int*)alloc(maxN * sizeof(int));
  int*   colidx  = (int*)alloc(maxE2 * sizeof(int));
  float* ebuf    = (float*)alloc(maxE2 * sizeof(float));
  float* part    = (float*)alloc(128 * 256 * sizeof(float));
  float* emb     = (float*)alloc(768 * sizeof(float));
  (void)ws_size; (void)out_size;

  // ---- encoders ----
  for (int s = 0; s < 3; ++s) {
    int nNodes = in_sizes[xIdx[s]] / Dd;
    int nEdges = in_sizes[eIdx[s]] / 2;
    const float* X0  = (const float*)d_in[xIdx[s]];
    const int*   ep  = (const int*)d_in[eIdx[s]];
    const int*   src = ep;
    const int*   dst = ep + nEdges;

    int gN = (nNodes + 255) / 256;
    int gE = (nEdges + 255) / 256;
    k_deg_init  <<<gN, 256, 0, stream>>>(fillpos, nNodes);
    k_deg_count <<<gE, 256, 0, stream>>>(dst, nEdges, fillpos);
    k_scan      <<<1, 1024, 0, stream>>>(fillpos, rowptr, nNodes);
    k_fill_self <<<gN, 256, 0, stream>>>(rowptr, fillpos, colidx, nNodes);
    k_fill_edges<<<gE, 256, 0, stream>>>(src, dst, nEdges, fillpos, colidx);

    const float* Xcur = X0;
    float* outs[2] = {bufA, bufB};
    for (int l = 0; l < 5; ++l) {
      int K = (l == 0) ? Dd : 256;
      int relu = (l != 4) ? 1 : 0;
      float* O = outs[l & 1];
      k_prep_wt<<<(256 * K + 255) / 256, 256, 0, stream>>>(enc[s].Wl[l], enc[s].Wr[l], Wlt, Wrt, K);
      k_gat_gemm<<<(nNodes + 15) / 16, 256, 0, stream>>>(Xcur, Wlt, Wrt, XL, XR, K, nNodes);
      k_gat_aggregate<<<(nNodes + 7) / 8, 256, 0, stream>>>(XL, XR, enc[s].att[l], enc[s].b[l],
                                                            rowptr, colidx, ebuf, O, nNodes, relu);
      Xcur = O;
    }
    k_pool_partial<<<128, 256, 0, stream>>>(Xcur, part, nNodes);
    k_pool_final  <<<1, 256, 0, stream>>>(part, emb + 256 * s, 128, 1.0f / (float)nNodes);
  }

  // ---- head ----
  k_mlp<<<1, 256, 0, stream>>>(emb, W1, B1, W2, B2, W3, B3, (float*)d_out);
}
